// GNN_SAGE_Model_86285892977273
// MI455X (gfx1250) — compile-verified, matrix operand order checked by reading, not run on hardware
//
#include <hip/hip_runtime.h>

// ---------------------------------------------------------------------------
// GraphSAGE 3-layer + head for MI455X (gfx1250, wave32, WMMA).
//  * GEMMs: v_wmma_f32_16x16x32_f16 (f16 A/B, f32 accumulate), 32-row x
//    512-col blocks, packed-B tiles loaded as contiguous b128 pairs.
//  * Aggregation: CSR (built once per call with int atomics) + register
//    gather fused into the GEMM's LDS staging -> zero float atomics.
//  * Working set (~45MB) is fully L2-resident (192MB L2); HBM irrelevant.
// ---------------------------------------------------------------------------

typedef __attribute__((ext_vector_type(16))) _Float16 v16h;
typedef __attribute__((ext_vector_type(8)))  float    v8f;

#define NNODES 10000
#define NEDGES 160000
#define HFEAT  512
#define OFEAT  128
#define KDIM   512

union F16Frag { uint4 u4[2]; v16h v; };   // 8 VGPRs = 16 halves
union H4 { _Float16 h[4]; uint2 u; };     // 4 halves = 8 bytes

// ---------------------------------------------------------------------------
__global__ __launch_bounds__(256) void fill_zero_i32(int* __restrict__ p, int n) {
  int i = blockIdx.x * 256 + threadIdx.x;
  if (i < n) p[i] = 0;
}

__global__ __launch_bounds__(256) void deg_kernel(const int* __restrict__ dst,
                                                  int* __restrict__ degi) {
  int e = blockIdx.x * 256 + threadIdx.x;
  if (e < NEDGES) atomicAdd(&degi[dst[e]], 1);
}

// Single-block exclusive scan over 10000 degree counts -> rowptr / cursor.
__global__ __launch_bounds__(1024) void scan_kernel(const int* __restrict__ degi,
                                                    int* __restrict__ rowptr,
                                                    int* __restrict__ cursor) {
  __shared__ int part[1024];
  const int tid  = threadIdx.x;
  const int base = tid * 10;               // 1024*10 >= 10000
  int local[10];
  int s = 0;
#pragma unroll
  for (int j = 0; j < 10; ++j) {
    int idx = base + j;
    int v = (idx < NNODES) ? degi[idx] : 0;
    local[j] = s;                          // exclusive within chunk
    s += v;
  }
  part[tid] = s;
  __syncthreads();
  for (int ofs = 1; ofs < 1024; ofs <<= 1) {   // Hillis-Steele inclusive scan
    int v   = part[tid];
    int add = (tid >= ofs) ? part[tid - ofs] : 0;
    __syncthreads();
    part[tid] = v + add;
    __syncthreads();
  }
  int chunk_off = (tid == 0) ? 0 : part[tid - 1];
#pragma unroll
  for (int j = 0; j < 10; ++j) {
    int idx = base + j;
    if (idx < NNODES) {
      int p = chunk_off + local[j];
      rowptr[idx] = p;
      cursor[idx] = p;
    }
  }
  if (tid == 1023) rowptr[NNODES] = part[1023];  // == NEDGES
}

__global__ __launch_bounds__(256) void fill_csr_kernel(const int* __restrict__ src,
                                                       const int* __restrict__ dst,
                                                       int* __restrict__ cursor,
                                                       int* __restrict__ csr_src) {
  int e = blockIdx.x * 256 + threadIdx.x;
  if (e < NEDGES) {
    int pos = atomicAdd(&cursor[dst[e]], 1);
    csr_src[pos] = src[e];
  }
}

// Repack fp32 row-major weight [K, Nout] into WMMA-B-native fp16 tiles.
// Tile (kk, nt): lane L -> n = nt*16 + (L&15), sel = L>>4; half h ->
// k = kk*32 + (h<8 ? h : h+8) + sel*8.  16 contiguous halves per lane.
__global__ __launch_bounds__(256) void pack_weights_kernel(const float* __restrict__ w,
                                                           _Float16* __restrict__ wp,
                                                           int K, int Nout) {
  int g = blockIdx.x * 256 + threadIdx.x;
  int numNT = Nout >> 4;
  int total = (K >> 5) * numNT * 32;
  if (g >= total) return;
  int lane = g & 31;
  int tile = g >> 5;
  int nt   = tile % numNT;
  int kk   = tile / numNT;
  int n    = nt * 16 + (lane & 15);
  int kb   = kk * 32 + (lane >> 4) * 8;
  F16Frag f;
#pragma unroll
  for (int h = 0; h < 16; ++h) {
    int k = kb + (h < 8 ? h : h + 8);
    f.v[h] = (_Float16)w[(size_t)k * Nout + n];
  }
  uint4* o = (uint4*)(wp + (size_t)g * 16);
  o[0] = f.u4[0];
  o[1] = f.u4[1];
}

// ---------------------------------------------------------------------------
// Fused SAGE layer:  out = act( mean_agg(xin) @ Wl + xin @ Wr + bias )
// Block: 256 thr (8 waves) -> 32 rows x 512 cols.  Each wave: 2 M-subtiles x
// 4 N-tiles (16 WMMAs per K-step, each B fragment reused twice).
// HAS_AGG=false: head layer, out = xin @ Wr + bias.
// ---------------------------------------------------------------------------
template <bool HAS_AGG>
__global__ __launch_bounds__(256) void sage_gemm_kernel(
    const int* __restrict__ rowptr,    // [N+1] (unused if !HAS_AGG)
    const int* __restrict__ csr_src,   // [E]   (unused if !HAS_AGG)
    const float* __restrict__ xin,     // [N,512] layer input (root + gather src)
    const _Float16* __restrict__ wlp,  // packed [512 x Nout] (unused if !HAS_AGG)
    const _Float16* __restrict__ wrp,  // packed [512 x Nout]
    const float* __restrict__ bias,    // [Nout]
    float* __restrict__ out,           // [N, Nout]
    int Nout, int do_relu) {
  __shared__ __align__(16) _Float16 lAroot[32 * KDIM];
  __shared__ __align__(16) _Float16 lAagg[HAS_AGG ? 32 * KDIM : 16];

  const int row0 = blockIdx.x * 32;
  const int tid  = threadIdx.x;

  // ---- Stage root rows: fp32 -> fp16 into LDS (float4-granular). ----
  for (int i = tid; i < 32 * 128; i += 256) {   // 32 rows x 128 float4
    int r = i >> 7, c4 = i & 127;
    int grow = row0 + r;
    float4 v = make_float4(0.f, 0.f, 0.f, 0.f);
    if (grow < NNODES) v = ((const float4*)(xin + (size_t)grow * KDIM))[c4];
    H4 h;
    h.h[0] = (_Float16)v.x; h.h[1] = (_Float16)v.y;
    h.h[2] = (_Float16)v.z; h.h[3] = (_Float16)v.w;
    *(uint2*)&lAroot[(size_t)r * KDIM + c4 * 4] = h.u;
  }

  // ---- Fused CSR mean-gather -> fp16 into LDS (no atomics). ----
  if constexpr (HAS_AGG) {
    for (int halfp = 0; halfp < 2; ++halfp) {
      int r    = halfp * 16 + (tid >> 4);       // 16 threads per row
      int grow = row0 + r;
      int c40  = tid & 15;                      // float4 col = c40 + j*16
      float4 acc[8];
#pragma unroll
      for (int j = 0; j < 8; ++j) acc[j] = make_float4(0.f, 0.f, 0.f, 0.f);
      float inv = 1.0f;
      if (grow < NNODES) {
        int e0 = rowptr[grow], e1 = rowptr[grow + 1];
        for (int e = e0; e < e1; ++e) {
          int s = csr_src[e];
          const float4* xs4 = (const float4*)(xin + (size_t)s * KDIM);
#pragma unroll
          for (int j = 0; j < 8; ++j) {
            float4 v = xs4[c40 + j * 16];
            acc[j].x += v.x; acc[j].y += v.y; acc[j].z += v.z; acc[j].w += v.w;
          }
        }
        int dg = e1 - e0;
        inv = 1.0f / (float)(dg > 1 ? dg : 1);
      }
#pragma unroll
      for (int j = 0; j < 8; ++j) {
        H4 h;
        h.h[0] = (_Float16)(acc[j].x * inv); h.h[1] = (_Float16)(acc[j].y * inv);
        h.h[2] = (_Float16)(acc[j].z * inv); h.h[3] = (_Float16)(acc[j].w * inv);
        *(uint2*)&lAagg[(size_t)r * KDIM + (c40 + j * 16) * 4] = h.u;
      }
    }
  }
  __syncthreads();

  // ---- WMMA compute ----
  const int wave = tid >> 5;
  const int lane = tid & 31;
  const int nb   = blockIdx.y * 512 + wave * 64;
  if (nb >= Nout) return;                       // head layer: waves 2..7 idle

  const int mloc  = lane & 15;
  const int sel   = lane >> 4;
  const int numNT = Nout >> 4;
  const int nt0   = nb >> 4;

  v8f acc[2][4];
#pragma unroll
  for (int ms = 0; ms < 2; ++ms)
#pragma unroll
    for (int t = 0; t < 4; ++t) acc[ms][t] = (v8f)(0.0f);

  const uint4* wl4 = (const uint4*)wlp;
  const uint4* wr4 = (const uint4*)wrp;

  for (int kk = 0; kk < KDIM / 32; ++kk) {
    const int ab0 = mloc * KDIM + kk * 32 + sel * 8;   // halves; 16B aligned
    const int ab1 = ab0 + 16 * KDIM;
    F16Frag aR0, aR1, aA0, aA1;
    aR0.u4[0] = *(const uint4*)(lAroot + ab0);
    aR0.u4[1] = *(const uint4*)(lAroot + ab0 + 16);
    aR1.u4[0] = *(const uint4*)(lAroot + ab1);
    aR1.u4[1] = *(const uint4*)(lAroot + ab1 + 16);
    if constexpr (HAS_AGG) {
      aA0.u4[0] = *(const uint4*)(lAagg + ab0);
      aA0.u4[1] = *(const uint4*)(lAagg + ab0 + 16);
      aA1.u4[0] = *(const uint4*)(lAagg + ab1);
      aA1.u4[1] = *(const uint4*)(lAagg + ab1 + 16);
    }
#pragma unroll
    for (int t = 0; t < 4; ++t) {
      int idx = (kk * numNT + (nt0 + t)) * 32 + lane;
      F16Frag br;
      br.u4[0] = wr4[2 * idx];
      br.u4[1] = wr4[2 * idx + 1];
      acc[0][t] = __builtin_amdgcn_wmma_f32_16x16x32_f16(false, aR0.v, false, br.v,
                                                         (short)0, acc[0][t], false, false);
      acc[1][t] = __builtin_amdgcn_wmma_f32_16x16x32_f16(false, aR1.v, false, br.v,
                                                         (short)0, acc[1][t], false, false);
      if constexpr (HAS_AGG) {
        F16Frag bl;
        bl.u4[0] = wl4[2 * idx];
        bl.u4[1] = wl4[2 * idx + 1];
        acc[0][t] = __builtin_amdgcn_wmma_f32_16x16x32_f16(false, aA0.v, false, bl.v,
                                                           (short)0, acc[0][t], false, false);
        acc[1][t] = __builtin_amdgcn_wmma_f32_16x16x32_f16(false, aA1.v, false, bl.v,
                                                           (short)0, acc[1][t], false, false);
      }
    }
  }

  // ---- Epilogue: bias + optional relu.  C/D: n = lane&15, m = v + 8*sel. ----
#pragma unroll
  for (int ms = 0; ms < 2; ++ms) {
#pragma unroll
    for (int t = 0; t < 4; ++t) {
      int n = nb + t * 16 + mloc;
      float bv = bias[n];
#pragma unroll
      for (int v = 0; v < 8; ++v) {
        int row = row0 + ms * 16 + sel * 8 + v;
        if (row < NNODES) {
          float r = acc[ms][t][v] + bv;
          if (do_relu) r = r > 0.0f ? r : 0.0f;
          out[(size_t)row * Nout + n] = r;
        }
      }
    }
  }
}

// ---------------------------------------------------------------------------
extern "C" void kernel_launch(void* const* d_in, const int* in_sizes, int n_in,
                              void* d_out, int out_size, void* d_ws, size_t ws_size,
                              hipStream_t stream) {
  const float* x    = (const float*)d_in[0];
  const int*   ei   = (const int*)d_in[1];   // (2,E): row0 = src, row1 = dst
  const float* wl0  = (const float*)d_in[2];
  const float* bl0  = (const float*)d_in[3];
  const float* wr0  = (const float*)d_in[4];
  const float* wl1  = (const float*)d_in[5];
  const float* bl1  = (const float*)d_in[6];
  const float* wr1  = (const float*)d_in[7];
  const float* wl2  = (const float*)d_in[8];
  const float* bl2  = (const float*)d_in[9];
  const float* wr2  = (const float*)d_in[10];
  const float* wout = (const float*)d_in[11];
  const float* bout = (const float*)d_in[12];
  const int* src = ei;
  const int* dst = ei + NEDGES;

  // Workspace layout (256B aligned), ~45MB total.
  char* ws = (char*)d_ws;
  size_t off = 0;
  auto take = [&](size_t bytes) -> void* {
    void* p = ws + off;
    off = (off + bytes + 255) & ~(size_t)255;
    return p;
  };
  const size_t NH = (size_t)NNODES * HFEAT;
  float* hA      = (float*)take(NH * sizeof(float));
  float* hB      = (float*)take(NH * sizeof(float));
  int*   degi    = (int*)take(NNODES * sizeof(int));
  int*   rowptr  = (int*)take((NNODES + 1) * sizeof(int));
  int*   cursor  = (int*)take(NNODES * sizeof(int));
  int*   csr_src = (int*)take(NEDGES * sizeof(int));
  _Float16* p_l0  = (_Float16*)take((size_t)512 * 512 * 2);
  _Float16* p_r0  = (_Float16*)take((size_t)512 * 512 * 2);
  _Float16* p_l1  = (_Float16*)take((size_t)512 * 512 * 2);
  _Float16* p_r1  = (_Float16*)take((size_t)512 * 512 * 2);
  _Float16* p_l2  = (_Float16*)take((size_t)512 * 512 * 2);
  _Float16* p_r2  = (_Float16*)take((size_t)512 * 512 * 2);
  _Float16* p_out = (_Float16*)take((size_t)512 * 128 * 2);

  // ---- Build CSR (once per call; only 320K int atomics). ----
  fill_zero_i32<<<(NNODES + 255) / 256, 256, 0, stream>>>(degi, NNODES);
  deg_kernel<<<(NEDGES + 255) / 256, 256, 0, stream>>>(dst, degi);
  scan_kernel<<<1, 1024, 0, stream>>>(degi, rowptr, cursor);
  fill_csr_kernel<<<(NEDGES + 255) / 256, 256, 0, stream>>>(src, dst, cursor, csr_src);

  // ---- Pack weights fp32 -> WMMA-native fp16 tiles. ----
  {
    int tot512 = (512 / 32) * (512 / 16) * 32;  // 16384 threads
    int tot128 = (512 / 32) * (128 / 16) * 32;  // 4096 threads
    pack_weights_kernel<<<(tot512 + 255) / 256, 256, 0, stream>>>(wl0, p_l0, 512, 512);
    pack_weights_kernel<<<(tot512 + 255) / 256, 256, 0, stream>>>(wr0, p_r0, 512, 512);
    pack_weights_kernel<<<(tot512 + 255) / 256, 256, 0, stream>>>(wl1, p_l1, 512, 512);
    pack_weights_kernel<<<(tot512 + 255) / 256, 256, 0, stream>>>(wr1, p_r1, 512, 512);
    pack_weights_kernel<<<(tot512 + 255) / 256, 256, 0, stream>>>(wl2, p_l2, 512, 512);
    pack_weights_kernel<<<(tot512 + 255) / 256, 256, 0, stream>>>(wr2, p_r2, 512, 512);
    pack_weights_kernel<<<(tot128 + 255) / 256, 256, 0, stream>>>(wout, p_out, 512, 128);
  }

  dim3 gemmGrid((NNODES + 31) / 32, 1);   // 313 blocks of 32 rows

  const float* layer_in[3]     = {x, hA, hB};
  float*       layer_out[3]    = {hA, hB, hA};
  const _Float16* layer_wl[3]  = {p_l0, p_l1, p_l2};
  const _Float16* layer_wr[3]  = {p_r0, p_r1, p_r2};
  const float* layer_b[3]      = {bl0, bl1, bl2};

  for (int l = 0; l < 3; ++l) {
    sage_gemm_kernel<true><<<gemmGrid, 256, 0, stream>>>(
        rowptr, csr_src, layer_in[l], layer_wl[l], layer_wr[l], layer_b[l],
        layer_out[l], HFEAT, /*relu=*/1);
  }

  // Head: out = h3 @ w_out + b_out (no aggregation, no relu).
  sage_gemm_kernel<false><<<gemmGrid, 256, 0, stream>>>(
      nullptr, nullptr, hA, nullptr, p_out, bout, (float*)d_out, OFEAT, /*relu=*/0);
}
